// GNNReID_49615462203952
// MI455X (gfx1250) — matrix operand order repaired and checked by reading, not compiled, not run on hardware
//
#include <hip/hip_runtime.h>

// ---------------- problem constants ----------------
#define NN      10000
#define DEG     16
#define EE      (NN*DEG)
#define DIN     2048
#define DD      512
#define NHEAD   8
#define HD      64
#define DHID    2048
#define NCLS    751
#define NCLS_P  768      // padded classifier rows (multiple of 16)

typedef __attribute__((ext_vector_type(16))) __bf16 v16bf;
typedef __attribute__((ext_vector_type(8)))  float  v8f;

union Frag16 { int4 q[2]; v16bf v; };   // 32 bytes = 16 bf16 = 8 VGPRs

__device__ __forceinline__ unsigned short f2bf(float f) {
    unsigned u = __float_as_uint(f);
    u += 0x7FFFu + ((u >> 16) & 1u);     // round-to-nearest-even
    return (unsigned short)(u >> 16);
}

// ---------------- f32 -> bf16 conversion (with zero pad) ----------------
__global__ void cvt_f32_bf16_pad(const float* __restrict__ src,
                                 unsigned short* __restrict__ dst,
                                 long n_src, long n_dst) {
    long i = (long)blockIdx.x * blockDim.x + threadIdx.x;
    long stride = (long)gridDim.x * blockDim.x;
    for (; i < n_dst; i += stride)
        dst[i] = (i < n_src) ? f2bf(src[i]) : (unsigned short)0;
}

// ---------------- WMMA GEMM: C[M,N] = A[M,K](bf16) @ W[N,K]^T(bf16) + bias ----------------
// Block = 256 threads = 8 waves, tiled 2(M) x 4(N); wave tile = 32 x 64
// (2 M-subtiles x 4 N-subtiles = 8 WMMAs per K-step of 32).
// Register double-buffered pipeline: K unrolled by 64, loads for the next
// K-step issue before the WMMAs of the current one.

__device__ __forceinline__ void load_frags(
    Frag16 (&a)[2], Frag16 (&b)[4],
    const unsigned short* __restrict__ arow0,
    const unsigned short* __restrict__ arow1,
    const unsigned short* __restrict__ wrow, int K, int k0)
{
    a[0].q[0] = *(const int4*)(arow0 + k0);        // K = k0 + lh*8 .. +7
    a[0].q[1] = *(const int4*)(arow0 + k0 + 16);   // K = k0 + 16 + lh*8 .. +7
    a[1].q[0] = *(const int4*)(arow1 + k0);
    a[1].q[1] = *(const int4*)(arow1 + k0 + 16);
    #pragma unroll
    for (int s = 0; s < 4; ++s) {
        const unsigned short* wp = wrow + (size_t)s * 16 * K + k0;
        b[s].q[0] = *(const int4*)(wp);            // K = k0 + lh*8 .. +7
        b[s].q[1] = *(const int4*)(wp + 16);       // K = k0 + 16 + lh*8 .. +7
    }
}

__device__ __forceinline__ void do_wmma(v8f (&acc)[2][4], Frag16 (&a)[2], Frag16 (&b)[4])
{
    #pragma unroll
    for (int mi = 0; mi < 2; ++mi)
        #pragma unroll
        for (int s = 0; s < 4; ++s)
            acc[mi][s] = __builtin_amdgcn_wmma_f32_16x16x32_bf16(
                false, a[mi].v, false, b[s].v, (short)0, acc[mi][s], false, false);
}

__global__ __launch_bounds__(256) void wmma_gemm_bf16(
    const unsigned short* __restrict__ A, const unsigned short* __restrict__ W,
    const float* __restrict__ bias, float* __restrict__ C,
    int M, int K, int Nout, int ldc, int relu)
{
    const int wave = threadIdx.x >> 5;
    const int lane = threadIdx.x & 31;
    const int wm   = wave & 1;                 // 2 M-waves (32 rows each)
    const int wn   = wave >> 1;                // 4 N-waves (64 cols each)
    const int m_base = blockIdx.x * 64 + wm * 32;
    const int n_base = blockIdx.y * 256 + wn * 64;
    if (m_base >= M) return;                   // M % 16 == 0; tile-level guards below

    const int l15 = lane & 15;
    const int lh  = lane >> 4;
    const bool m1_valid = (m_base + 16) < M;   // second 16-row subtile in range?

    // A fragment rows (16-bit A 16x32 layout): row = base + (lane&15),
    // K halves selected by lane>>4. Clamp invalid subtile to a valid row.
    const unsigned short* arow0 = A + (size_t)(m_base + l15) * K + lh * 8;
    const unsigned short* arow1 = m1_valid ? (arow0 + (size_t)16 * K) : arow0;
    // B fragment: column n = n_base + s*16 + (lane&15); K halves selected by lane>>4 (same as A)
    const unsigned short* wrow  = W + (size_t)(n_base + l15) * K + lh * 8;

    v8f acc[2][4];
    #pragma unroll
    for (int mi = 0; mi < 2; ++mi)
        #pragma unroll
        for (int s = 0; s < 4; ++s)
            #pragma unroll
            for (int i = 0; i < 8; ++i) acc[mi][s][i] = 0.0f;

    Frag16 a0[2], b0[4], a1[2], b1[4];

    // --- pipelined main loop (K % 64 == 0 for every GEMM here) ---
    load_frags(a0, b0, arow0, arow1, wrow, K, 0);
    int k0 = 0;
    for (; k0 + 64 < K; k0 += 64) {
        load_frags(a1, b1, arow0, arow1, wrow, K, k0 + 32);   // prefetch stage 1
        do_wmma(acc, a0, b0);
        load_frags(a0, b0, arow0, arow1, wrow, K, k0 + 64);   // prefetch stage 0
        do_wmma(acc, a1, b1);
    }
    load_frags(a1, b1, arow0, arow1, wrow, K, k0 + 32);
    do_wmma(acc, a0, b0);
    do_wmma(acc, a1, b1);

    // --- epilogue: D layout VGPR i -> M = (lane>>4)*8 + i, N = lane&15 ---
    #pragma unroll
    for (int mi = 0; mi < 2; ++mi) {
        if (mi == 1 && !m1_valid) break;
        const int row0 = m_base + mi * 16 + lh * 8;
        #pragma unroll
        for (int s = 0; s < 4; ++s) {
            const int col = n_base + s * 16 + l15;
            if (col < Nout) {
                const float bs = bias ? bias[col] : 0.0f;
                float* cp = C + (size_t)row0 * ldc + col;
                #pragma unroll
                for (int i = 0; i < 8; ++i) {
                    float val = acc[mi][s][i] + bs;
                    if (relu) val = fmaxf(val, 0.0f);
                    cp[(size_t)i * ldc] = val;
                }
            }
        }
    }
}

// ---------------- edge attention + aggregation ----------------
// c = repeat(arange(N), DEG): destination node i owns edges [i*DEG, i*DEG+DEG).
// One wave per (node, head): dot via wave reduction, width-16 softmax, float2 agg.
__global__ __launch_bounds__(256) void attn_agg(
    const float* __restrict__ q, const float* __restrict__ k,
    const float* __restrict__ v, const int* __restrict__ edge_index,
    float* __restrict__ agg)
{
    const int task = blockIdx.x * 8 + (threadIdx.x >> 5);
    const int lane = threadIdx.x & 31;
    if (task >= NN * NHEAD) return;
    const int node = task >> 3;      // / NHEAD
    const int h    = task & 7;
    const int base = node * DD + h * HD + lane * 2;

    const float2 qv = *(const float2*)(q + base);
    int r_l = 0;
    if (lane < DEG) r_l = edge_index[(size_t)(node * DEG + lane) * 2];  // source r

    float myscore = 0.0f;
    #pragma unroll
    for (int j = 0; j < DEG; ++j) {
        const int rj = __shfl(r_l, j, 32);
        const float2 kv = *(const float2*)(k + rj * DD + h * HD + lane * 2);
        float p = qv.x * kv.x + qv.y * kv.y;
        #pragma unroll
        for (int m = 16; m >= 1; m >>= 1) p += __shfl_xor(p, m, 32);
        if (lane == j) myscore = p * 0.125f;          // 1/sqrt(HD)
    }
    // softmax over the 16 edges (lanes 0..15 hold scores; width-16 shuffles)
    float mx = myscore;
    #pragma unroll
    for (int m = 8; m >= 1; m >>= 1) mx = fmaxf(mx, __shfl_xor(mx, m, 16));
    const float ex = __expf(myscore - mx);
    float den = ex;
    #pragma unroll
    for (int m = 8; m >= 1; m >>= 1) den += __shfl_xor(den, m, 16);
    const float alpha = ex / den;

    float2 acc = make_float2(0.f, 0.f);
    #pragma unroll
    for (int j = 0; j < DEG; ++j) {
        const float a = __shfl(alpha, j, 32);
        const int  rj = __shfl(r_l, j, 32);
        const float2 vv = *(const float2*)(v + rj * DD + h * HD + lane * 2);
        acc.x += a * vv.x; acc.y += a * vv.y;
    }
    *(float2*)(agg + base) = acc;
}

// ---------------- residual + LayerNorm (wave per row, D = 512) ----------------
__global__ __launch_bounds__(256) void ln_residual(
    const float* __restrict__ x, const float* __restrict__ f2,
    const float* __restrict__ g, const float* __restrict__ b,
    float* __restrict__ out)
{
    const int row  = blockIdx.x * 8 + (threadIdx.x >> 5);
    const int lane = threadIdx.x & 31;
    if (row >= NN) return;
    const float* xr = x  + (size_t)row * DD;
    const float* fr = f2 + (size_t)row * DD;
    float vals[16];
    float s = 0.f, s2 = 0.f;
    #pragma unroll
    for (int t = 0; t < 16; ++t) {
        const float u = xr[lane + t * 32] + fr[lane + t * 32];
        vals[t] = u; s += u; s2 += u * u;
    }
    #pragma unroll
    for (int m = 16; m >= 1; m >>= 1) { s += __shfl_xor(s, m, 32); s2 += __shfl_xor(s2, m, 32); }
    const float mean = s * (1.0f / DD);
    const float var  = s2 * (1.0f / DD) - mean * mean;
    const float inv  = rsqrtf(var + 1e-5f);
    float* orow = out + (size_t)row * DD;
    #pragma unroll
    for (int t = 0; t < 16; ++t) {
        const int c = lane + t * 32;
        orow[c] = (vals[t] - mean) * inv * g[c] + b[c];
    }
}

// ---------------- final: feats_out copy + bn-scale + bf16 convert ----------------
__global__ void bn_cvt(const float* __restrict__ x, const float* __restrict__ g,
                       const float* __restrict__ b, float* __restrict__ feats_out,
                       unsigned short* __restrict__ bnbf)
{
    const float rs = rsqrtf(1.0f + 1e-5f);
    long i = (long)blockIdx.x * blockDim.x + threadIdx.x;
    const long stride = (long)gridDim.x * blockDim.x;
    for (; i < (long)NN * DD; i += stride) {
        const float val = x[i];
        const int col = (int)(i % DD);
        feats_out[i] = val;
        bnbf[i] = f2bf(val * rs * g[col] + b[col]);
    }
}

// ---------------- host orchestration ----------------
extern "C" void kernel_launch(void* const* d_in, const int* in_sizes, int n_in,
                              void* d_out, int out_size, void* d_ws, size_t ws_size,
                              hipStream_t stream) {
    const float* feats = (const float*)d_in[0];
    const int*   eidx  = (const int*)  d_in[1];
    const float* Wred  = (const float*)d_in[3];
    const float* bred  = (const float*)d_in[4];
    const float* Wq = (const float*)d_in[5];   const float* bq = (const float*)d_in[6];
    const float* Wk = (const float*)d_in[7];   const float* bk = (const float*)d_in[8];
    const float* Wv = (const float*)d_in[9];   const float* bv = (const float*)d_in[10];
    const float* Wo = (const float*)d_in[11];  const float* bo = (const float*)d_in[12];
    const float* g1 = (const float*)d_in[13];  const float* be1 = (const float*)d_in[14];
    const float* W1 = (const float*)d_in[15];  const float* b1 = (const float*)d_in[16];
    const float* W2 = (const float*)d_in[17];  const float* b2 = (const float*)d_in[18];
    const float* g2 = (const float*)d_in[19];  const float* be2 = (const float*)d_in[20];
    const float* bn_g = (const float*)d_in[21];
    const float* bn_b = (const float*)d_in[22];
    const float* Wfc  = (const float*)d_in[23];

    // ---- workspace carve-up (256B aligned) ----
    char* ws = (char*)d_ws; size_t off = 0;
    auto carve = [&](size_t bytes) -> char* {
        char* p = ws + off; off += (bytes + 255) & ~(size_t)255; return p;
    };
    unsigned short* wred_bf = (unsigned short*)carve((size_t)DD  * DIN  * 2);
    unsigned short* wq_bf   = (unsigned short*)carve((size_t)2 * DD * DD * 2);
    unsigned short* wk_bf   = (unsigned short*)carve((size_t)2 * DD * DD * 2);
    unsigned short* wv_bf   = (unsigned short*)carve((size_t)2 * DD * DD * 2);
    unsigned short* wo_bf   = (unsigned short*)carve((size_t)2 * DD * DD * 2);
    unsigned short* w1_bf   = (unsigned short*)carve((size_t)2 * DHID * DD * 2);
    unsigned short* w2_bf   = (unsigned short*)carve((size_t)2 * DD * DHID * 2);
    unsigned short* wfc_bf  = (unsigned short*)carve((size_t)NCLS_P * DD * 2);
    unsigned short* xbf     = (unsigned short*)carve((size_t)NN * DIN * 2); // reused for agg/hid/bn bf16
    float* x    = (float*)carve((size_t)NN * DD * 4);
    float* qb   = (float*)carve((size_t)NN * DD * 4);
    float* kb   = (float*)carve((size_t)NN * DD * 4);
    float* vb   = (float*)carve((size_t)NN * DD * 4);
    float* agg  = (float*)carve((size_t)NN * DD * 4);
    float* f2b  = (float*)carve((size_t)NN * DD * 4);
    float* hid  = (float*)carve((size_t)NN * DHID * 4);

    float* logits    = (float*)d_out;
    float* feats_out = (float*)d_out + (size_t)NN * NCLS;

    const dim3 blk(256);
    auto cvt = [&](const float* s, unsigned short* d, long nsrc, long ndst) {
        cvt_f32_bf16_pad<<<2048, blk, 0, stream>>>(s, d, nsrc, ndst);
    };
    auto gemm = [&](const unsigned short* A, const unsigned short* W, const float* bias,
                    float* C, int M, int K, int Nout, int Npad, int ldc, int relu) {
        dim3 grid((M + 63) / 64, (Npad + 255) / 256);
        wmma_gemm_bf16<<<grid, blk, 0, stream>>>(A, W, bias, C, M, K, Nout, ldc, relu);
    };

    // 1. weights -> bf16 (Wfc zero-padded to 768 rows)
    cvt(Wred, wred_bf, (long)DD * DIN,      (long)DD * DIN);
    cvt(Wq,   wq_bf,   (long)2 * DD * DD,   (long)2 * DD * DD);
    cvt(Wk,   wk_bf,   (long)2 * DD * DD,   (long)2 * DD * DD);
    cvt(Wv,   wv_bf,   (long)2 * DD * DD,   (long)2 * DD * DD);
    cvt(Wo,   wo_bf,   (long)2 * DD * DD,   (long)2 * DD * DD);
    cvt(W1,   w1_bf,   (long)2 * DHID * DD, (long)2 * DHID * DD);
    cvt(W2,   w2_bf,   (long)2 * DD * DHID, (long)2 * DD * DHID);
    cvt(Wfc,  wfc_bf,  (long)NCLS * DD,     (long)NCLS_P * DD);

    // 2. x = feats @ Wred^T + bred
    cvt(feats, xbf, (long)NN * DIN, (long)NN * DIN);
    gemm(xbf, wred_bf, bred, x, NN, DIN, DD, DD, DD, 0);

    for (int i = 0; i < 2; ++i) {
        const unsigned short* wqi = wq_bf + (size_t)i * DD * DD;
        const unsigned short* wki = wk_bf + (size_t)i * DD * DD;
        const unsigned short* wvi = wv_bf + (size_t)i * DD * DD;
        const unsigned short* woi = wo_bf + (size_t)i * DD * DD;
        const unsigned short* w1i = w1_bf + (size_t)i * DHID * DD;
        const unsigned short* w2i = w2_bf + (size_t)i * DD * DHID;

        // 3. q/k/v projections
        cvt(x, xbf, (long)NN * DD, (long)NN * DD);
        gemm(xbf, wqi, bq + i * DD, qb, NN, DD, DD, DD, DD, 0);
        gemm(xbf, wki, bk + i * DD, kb, NN, DD, DD, DD, DD, 0);
        gemm(xbf, wvi, bv + i * DD, vb, NN, DD, DD, DD, DD, 0);

        // 4. edge attention + aggregation
        attn_agg<<<(NN * NHEAD) / 8, blk, 0, stream>>>(qb, kb, vb, eidx, agg);

        // 5. f2 = agg @ Wo^T + bo ; x = LN(x + f2)
        cvt(agg, xbf, (long)NN * DD, (long)NN * DD);
        gemm(xbf, woi, bo + i * DD, f2b, NN, DD, DD, DD, DD, 0);
        ln_residual<<<NN / 8, blk, 0, stream>>>(x, f2b, g1 + i * DD, be1 + i * DD, x);

        // 6. MLP: hid = relu(x @ W1^T + b1); f2 = hid @ W2^T + b2; x = LN(x + f2)
        cvt(x, xbf, (long)NN * DD, (long)NN * DD);
        gemm(xbf, w1i, b1 + i * DHID, hid, NN, DD, DHID, DHID, DHID, 1);
        cvt(hid, xbf, (long)NN * DHID, (long)NN * DHID);
        gemm(xbf, w2i, b2 + i * DD, f2b, NN, DHID, DD, DD, DD, 0);
        ln_residual<<<NN / 8, blk, 0, stream>>>(x, f2b, g2 + i * DD, be2 + i * DD, x);
    }

    // 7. feats_out = x ; bn = x * rsqrt(1+eps) * bn_g + bn_b (bf16) ; logits = bn @ Wfc^T
    bn_cvt<<<2048, blk, 0, stream>>>(x, bn_g, bn_b, feats_out, xbf);
    gemm(xbf, wfc_bf, nullptr, logits, NN, DD, NCLS, NCLS_P, NCLS, 0);
}
